// LSTM_18889266168155
// MI455X (gfx1250) — matrix operand order repaired and checked by reading, not compiled, not run on hardware
//
#include <hip/hip_runtime.h>

// Problem constants (from reference): B=256, T=512, D=64, H=256, L=2, FC=128, OUT=1
#define Bq 256
#define Tq 512
#define Dq 64
#define Hq 256
#define FCq 128
#define NBLK 32   // persistent blocks per layer kernel (co-resident on MI455X)

typedef _Float16 v16h __attribute__((ext_vector_type(16)));
typedef _Float16 v8h  __attribute__((ext_vector_type(8)));
typedef float    v8f  __attribute__((ext_vector_type(8)));

// ---------------- prep kernels ----------------

__global__ void k_f32_to_f16(const float* __restrict__ s, _Float16* __restrict__ d, int n) {
    int i = blockIdx.x * blockDim.x + threadIdx.x;
    if (i < n) d[i] = (_Float16)s[i];
}

__global__ void k_bias_sum(const float* __restrict__ a, const float* __restrict__ b,
                           float* __restrict__ o, int n) {
    int i = blockIdx.x * blockDim.x + threadIdx.x;
    if (i < n) o[i] = a[i] + b[i];
}

__global__ void k_zero_u32(unsigned int* __restrict__ p, int nwords) {
    int i = blockIdx.x * blockDim.x + threadIdx.x;
    if (i < nwords) p[i] = 0u;
}

// ---------------- device-wide barrier (monotone counter, 32 co-resident blocks) ----------------

__device__ __forceinline__ void grid_sync(unsigned int* bar, unsigned int target) {
    __syncthreads();
    if (threadIdx.x == 0) {
        __threadfence();  // make this block's h/c/y stores device-visible
        __hip_atomic_fetch_add(bar, 1u, __ATOMIC_RELEASE, __HIP_MEMORY_SCOPE_AGENT);
        while (__hip_atomic_load(bar, __ATOMIC_ACQUIRE, __HIP_MEMORY_SCOPE_AGENT) < target) {
            __builtin_amdgcn_s_sleep(1);
        }
    }
    __syncthreads();
}

// ---------------- persistent LSTM layer kernel ----------------
//
// Grid: NBLK=32 blocks x 256 threads = 256 waves (wave32), resident for the
// whole sequence. Block b owns hidden tile nT = b>>1 and batch tiles
// mT = (b&1)*8 .. +7 (one per wave). All 8 waves of a block share identical
// B fragments (the 4 gate-weight tiles for nT), so those are staged in LDS
// once and reused for all 512 timesteps:
//   sW[0            .. 64*Din)  : Wih rows (g*H + n0 + c), K-major
//   sW[64*Din .. 64*Din+64*H)   : Whh rows (g*H + n0 + c), K-major
// Per step, wave (mT,nT) computes the 16x16 (batch x hidden) block for all
// four gates with v_wmma_f32_16x16x32_f16 and does the cell update in-register.
// h is double-buffered in f16; a device-wide barrier separates steps.

__global__ __launch_bounds__(256, 1)
void k_lstm_layer(const _Float16* __restrict__ xin,   // layer input, [B, T, Din] f16
                  int xin_stride, int Din,             // row stride = T*Din
                  const _Float16* __restrict__ Wih,    // [4H, Din] f16
                  const _Float16* __restrict__ Whh,    // [4H, H]   f16
                  const float*    __restrict__ bias,   // [4H] = b_ih + b_hh
                  _Float16* __restrict__ h16a,         // [B, H] f16 (t even: read)
                  _Float16* __restrict__ h16b,         // [B, H] f16 (t even: write)
                  float*    __restrict__ h32,          // [B, H] f32 final hidden
                  float*    __restrict__ c32,          // [B, H] f32 cell state
                  _Float16* __restrict__ yout,         // [B, T, H] f16 or nullptr
                  unsigned int* __restrict__ bar)      // barrier counter (pre-zeroed)
{
    __shared__ __align__(32) _Float16 sW[64 * (Hq + Hq)];  // 64 KB max (Din<=H)

    const int tid  = threadIdx.x;
    const int lane = tid & 31;
    const int wl   = tid >> 5;                     // wave-in-block 0..7
    const int nT   = blockIdx.x >> 1;              // hidden tile 0..15
    const int mT   = ((blockIdx.x & 1) << 3) + wl; // batch tile 0..15
    const int r    = lane & 15;
    const int kh   = lane >> 4;
    const int b0   = mT * 16;
    const int n0   = nT * 16;
    const int whhOff = 64 * Din;                   // LDS offset of Whh stage

    // ---- stage this block's gate-weight tiles into LDS (once) ----
    for (int idx = tid; idx < 64 * (Din / 8); idx += 256) {
        int rowc = idx / (Din / 8);                // g*16 + c
        int kk   = (idx % (Din / 8)) * 8;
        int g = rowc >> 4, c = rowc & 15;
        *(v8h*)&sW[rowc * Din + kk] =
            *(const v8h*)&Wih[(size_t)(g * Hq + n0 + c) * Din + kk];
    }
    for (int idx = tid; idx < 64 * (Hq / 8); idx += 256) {
        int rowc = idx / (Hq / 8);
        int kk   = (idx % (Hq / 8)) * 8;
        int g = rowc >> 4, c = rowc & 15;
        *(v8h*)&sW[whhOff + rowc * Hq + kk] =
            *(const v8h*)&Whh[(size_t)(g * Hq + n0 + c) * Hq + kk];
    }
    __syncthreads();

    const int   hidx = n0 + r;
    const float bi = bias[hidx];
    const float bf = bias[Hq + hidx];
    const float bg = bias[2 * Hq + hidx];
    const float bo = bias[3 * Hq + hidx];

    for (int t = 0; t < Tq; ++t) {
        const _Float16* hp = (t & 1) ? h16b : h16a;
        _Float16*       hn = (t & 1) ? h16a : h16b;

        v8f acc[4] = {};   // i, f, g, o accumulators (fp32)

        // ---- input contribution: A = x_t tile (global), B = Wih^T tile (LDS) ----
        {
            const _Float16* aA = xin + (size_t)(b0 + r) * xin_stride + t * Din;
            for (int kc = 0; kc < Din; kc += 32) {
                // A 16x32 layout (contiguous, matches B): lanes 0-15 K=[0..15];
                // lanes 16-31 K=[16..31]
                v8h alo = *(const v8h*)(aA + kc + kh * 16);
                v8h ahi = *(const v8h*)(aA + kc + kh * 16 + 8);
                v16h a  = __builtin_shufflevector(alo, ahi,
                           0,1,2,3,4,5,6,7,8,9,10,11,12,13,14,15);
#pragma unroll
                for (int g = 0; g < 4; ++g) {
                    v16h bw = *(const v16h*)&sW[(g * 16 + r) * Din + kc + kh * 16];
                    acc[g] = __builtin_amdgcn_wmma_f32_16x16x32_f16(
                        false, a, false, bw, (short)0, acc[g], false, false);
                }
            }
        }

        // ---- recurrent contribution: A = h_{t-1} tile (global), B = Whh^T tile (LDS) ----
        {
            const _Float16* hA = hp + (size_t)(b0 + r) * Hq;
#pragma unroll 2
            for (int kc = 0; kc < Hq; kc += 32) {
                v8h alo = *(const v8h*)(hA + kc + kh * 16);
                v8h ahi = *(const v8h*)(hA + kc + kh * 16 + 8);
                v16h a  = __builtin_shufflevector(alo, ahi,
                           0,1,2,3,4,5,6,7,8,9,10,11,12,13,14,15);
#pragma unroll
                for (int g = 0; g < 4; ++g) {
                    v16h bw = *(const v16h*)&sW[whhOff + (g * 16 + r) * Hq + kc + kh * 16];
                    acc[g] = __builtin_amdgcn_wmma_f32_16x16x32_f16(
                        false, a, false, bw, (short)0, acc[g], false, false);
                }
            }
        }

        // ---- LSTM cell pointwise update (PyTorch gate order i,f,g,o) ----
        // C/D layout: vgpr v: lanes 0-15 -> row v, lanes 16-31 -> row v+8; col = lane&15
#pragma unroll
        for (int v = 0; v < 8; ++v) {
            const int    b  = b0 + v + 8 * kh;
            const size_t ci = (size_t)b * Hq + hidx;
            float iv = 1.0f / (1.0f + expf(-(acc[0][v] + bi)));
            float fv = 1.0f / (1.0f + expf(-(acc[1][v] + bf)));
            float gv = tanhf(acc[2][v] + bg);
            float ov = 1.0f / (1.0f + expf(-(acc[3][v] + bo)));
            float cn = fv * c32[ci] + iv * gv;
            float hn_ = ov * tanhf(cn);
            c32[ci] = cn;
            h32[ci] = hn_;
            hn[ci]  = (_Float16)hn_;
            if (yout) yout[(size_t)b * (Tq * Hq) + t * Hq + hidx] = (_Float16)hn_;
        }

        // ---- device-wide step barrier ----
        grid_sync(bar, (unsigned int)(t + 1) * NBLK);
    }
}

// ---------------- MLP head: out = relu(relu(h_n) @ fc1^T + b1) @ fc2^T + b2 ----------------

__global__ __launch_bounds__(FCq)
void k_mlp(const float* __restrict__ h0, const float* __restrict__ h1,
           const float* __restrict__ fc1w, const float* __restrict__ fc1b,
           const float* __restrict__ fc2w, const float* __restrict__ fc2b,
           float* __restrict__ out)
{
    __shared__ float red[FCq];
    const int i = blockIdx.x;                  // 0..511
    const int j = threadIdx.x;                 // 0..127
    const float* hr = (i < Bq) ? (h0 + (size_t)i * Hq) : (h1 + (size_t)(i - Bq) * Hq);
    float s = fc1b[j];
    for (int k = 0; k < Hq; ++k)
        s += fmaxf(hr[k], 0.0f) * fc1w[(size_t)j * Hq + k];
    red[j] = fmaxf(s, 0.0f) * fc2w[j];
    __syncthreads();
    for (int off = FCq / 2; off > 0; off >>= 1) {
        if (j < off) red[j] += red[j + off];
        __syncthreads();
    }
    if (j == 0) out[i] = red[0] + fc2b[0];
}

// ---------------- host-side launch ----------------

extern "C" void kernel_launch(void* const* d_in, const int* in_sizes, int n_in,
                              void* d_out, int out_size, void* d_ws, size_t ws_size,
                              hipStream_t stream) {
    const float* x    = (const float*)d_in[0];
    const float* Wih0 = (const float*)d_in[1];
    const float* Whh0 = (const float*)d_in[2];
    const float* bih0 = (const float*)d_in[3];
    const float* bhh0 = (const float*)d_in[4];
    const float* Wih1 = (const float*)d_in[5];
    const float* Whh1 = (const float*)d_in[6];
    const float* bih1 = (const float*)d_in[7];
    const float* bhh1 = (const float*)d_in[8];
    const float* fc1w = (const float*)d_in[9];
    const float* fc1b = (const float*)d_in[10];
    const float* fc2w = (const float*)d_in[11];
    const float* fc2b = (const float*)d_in[12];
    float* out = (float*)d_out;

    // ---- workspace carve-up (256B aligned) ----
    char*  base = (char*)d_ws;
    size_t off  = 0;
    auto alloc = [&](size_t bytes) -> char* {
        char* p = base + off;
        off = (off + bytes + 255) & ~(size_t)255;
        return p;
    };
    _Float16* Wih0h = (_Float16*)alloc((size_t)4 * Hq * Dq * 2);
    _Float16* Whh0h = (_Float16*)alloc((size_t)4 * Hq * Hq * 2);
    _Float16* Wih1h = (_Float16*)alloc((size_t)4 * Hq * Hq * 2);
    _Float16* Whh1h = (_Float16*)alloc((size_t)4 * Hq * Hq * 2);
    float*    bias0 = (float*)alloc((size_t)4 * Hq * 4);
    float*    bias1 = (float*)alloc((size_t)4 * Hq * 4);
    _Float16* xh    = (_Float16*)alloc((size_t)Bq * Tq * Dq * 2);  // 16.8 MB
    _Float16* y0h   = (_Float16*)alloc((size_t)Bq * Tq * Hq * 2);  // 67 MB
    _Float16* h16a0 = (_Float16*)alloc((size_t)Bq * Hq * 2);
    _Float16* h16b0 = (_Float16*)alloc((size_t)Bq * Hq * 2);
    _Float16* h16a1 = (_Float16*)alloc((size_t)Bq * Hq * 2);
    _Float16* h16b1 = (_Float16*)alloc((size_t)Bq * Hq * 2);
    float*    h32_0 = (float*)alloc((size_t)Bq * Hq * 4);
    float*    c32_0 = (float*)alloc((size_t)Bq * Hq * 4);
    float*    h32_1 = (float*)alloc((size_t)Bq * Hq * 4);
    float*    c32_1 = (float*)alloc((size_t)Bq * Hq * 4);
    unsigned int* bars = (unsigned int*)alloc(2 * sizeof(unsigned int));
    (void)ws_size; (void)in_sizes; (void)n_in; (void)out_size;

    // ---- prep: conversions, bias fusion, state + barrier zeroing ----
    auto cvt = [&](const float* s, _Float16* d, int n) {
        k_f32_to_f16<<<(n + 255) / 256, 256, 0, stream>>>(s, d, n);
    };
    cvt(Wih0, Wih0h, 4 * Hq * Dq);
    cvt(Whh0, Whh0h, 4 * Hq * Hq);
    cvt(Wih1, Wih1h, 4 * Hq * Hq);
    cvt(Whh1, Whh1h, 4 * Hq * Hq);
    cvt(x, xh, Bq * Tq * Dq);
    k_bias_sum<<<(4 * Hq + 255) / 256, 256, 0, stream>>>(bih0, bhh0, bias0, 4 * Hq);
    k_bias_sum<<<(4 * Hq + 255) / 256, 256, 0, stream>>>(bih1, bhh1, bias1, 4 * Hq);
    auto zero = [&](void* p, size_t bytes) {
        int nw = (int)(bytes / 4);
        k_zero_u32<<<(nw + 255) / 256, 256, 0, stream>>>((unsigned int*)p, nw);
    };
    zero(h16a0, (size_t)Bq * Hq * 2); zero(h16b0, (size_t)Bq * Hq * 2);
    zero(h16a1, (size_t)Bq * Hq * 2); zero(h16b1, (size_t)Bq * Hq * 2);
    zero(c32_0, (size_t)Bq * Hq * 4); zero(c32_1, (size_t)Bq * Hq * 4);
    zero(h32_0, (size_t)Bq * Hq * 4); zero(h32_1, (size_t)Bq * Hq * 4);
    zero(bars, 2 * sizeof(unsigned int));

    // ---- layer 0: one persistent kernel over all 512 steps ----
    k_lstm_layer<<<NBLK, 256, 0, stream>>>(
        xh, Tq * Dq, Dq, Wih0h, Whh0h, bias0,
        h16a0, h16b0, h32_0, c32_0, y0h, &bars[0]);

    // ---- layer 1: one persistent kernel over all 512 steps ----
    k_lstm_layer<<<NBLK, 256, 0, stream>>>(
        y0h, Tq * Hq, Hq, Wih1h, Whh1h, bias1,
        h16a1, h16b1, h32_1, c32_1, nullptr, &bars[1]);

    // ---- MLP head on stacked final hidden states [2B, H] -> [2B, 1] ----
    k_mlp<<<2 * Bq, FCq, 0, stream>>>(h32_0, h32_1, fc1w, fc1b, fc2w, fc2b, out);
}
